// RGCN_41266045780248
// MI455X (gfx1250) — compile-verified, matrix operand order checked by reading, not run on hardware
//
#include <hip/hip_runtime.h>
#include <stdint.h>

// Problem constants (match reference)
#define BB 4
#define NN 4096
#define EE 32768
#define DD 768
#define RR 3
#define NG 4          // RR relations + root folded in as g==3
#define NK (DD / 32)  // 24 K-steps of 32

typedef __attribute__((ext_vector_type(16))) __bf16 v16bf;
typedef __attribute__((ext_vector_type(8)))  float  v8f;
typedef __attribute__((ext_vector_type(4)))  unsigned int u32x4;
typedef __attribute__((ext_vector_type(2)))  unsigned int u32x2;
typedef __attribute__((ext_vector_type(4)))  float        f32x4;
typedef __attribute__((ext_vector_type(4)))  int          i32x4;
typedef __attribute__((ext_vector_type(8)))  int          i32x8;

// Explicit address-space pointers with *native* vector types (HIP_vector_type
// classes cannot copy across address spaces): force global_load_/global_store_
typedef __attribute__((address_space(1))) const u32x4*  GQc;
typedef __attribute__((address_space(1)))       float*  GFp;
typedef __attribute__((address_space(1))) const f32x4*  GF4c;
typedef __attribute__((address_space(1)))       u32x2*  GU2p;
typedef __attribute__((address_space(3))) const u32x4*  LQc;
typedef __attribute__((address_space(3)))       u32x4*  LQp;
typedef __attribute__((address_space(3))) unsigned short* LS16;

#if defined(__has_builtin)
#if __has_builtin(__builtin_amdgcn_tensor_load_to_lds) && __has_builtin(__builtin_amdgcn_s_wait_tensorcnt)
#define HAVE_TDM 1
#endif
#endif
#ifndef HAVE_TDM
#define HAVE_TDM 0
#endif

union Frag { v16bf v; u32x4 q[2]; };

__device__ __forceinline__ unsigned short f32_to_bf16(float f) {
    union { float f; unsigned u; } c; c.f = f;
    unsigned u = c.u;
    unsigned r = u + 0x7FFFu + ((u >> 16) & 1u);   // round-to-nearest-even
    if ((u & 0x7F800000u) == 0x7F800000u) r = u;   // preserve NaN/Inf
    return (unsigned short)(r >> 16);
}

// ---------------------------------------------------------------- utilities
__global__ void zero_f32(float* __restrict__ p, int n) {
    int i = blockIdx.x * blockDim.x + threadIdx.x;
    if (i < n) p[i] = 0.0f;
}

// f32 -> bf16 (optionally fused ReLU), 4 elements per thread
__global__ void convert_bf16(const float* __restrict__ in,
                             unsigned short* __restrict__ out,
                             int n4, int relu) {
    int i = blockIdx.x * blockDim.x + threadIdx.x;
    if (i >= n4) return;
    f32x4 f = ((GF4c)(const f32x4*)in)[i];
    if (relu) {
        f.x = fmaxf(f.x, 0.0f); f.y = fmaxf(f.y, 0.0f);
        f.z = fmaxf(f.z, 0.0f); f.w = fmaxf(f.w, 0.0f);
    }
    u32x2 o;
    o.x = (unsigned)f32_to_bf16(f.x) | ((unsigned)f32_to_bf16(f.y) << 16);
    o.y = (unsigned)f32_to_bf16(f.z) | ((unsigned)f32_to_bf16(f.w) << 16);
    ((GU2p)(u32x2*)out)[i] = o;
}

// wp[g][n_out][k] = bf16( src_g[k][n_out] )   (transposed pack -> contiguous K)
__global__ void pack_weights(const float* __restrict__ W,
                             const float* __restrict__ root,
                             unsigned short* __restrict__ wp) {
    int i = blockIdx.x * blockDim.x + threadIdx.x;
    if (i >= NG * DD * DD) return;
    int k = i % DD;
    int n = (i / DD) % DD;
    int g = i / (DD * DD);
    const float* src = (g < RR) ? (W + (size_t)g * DD * DD) : root;
    wp[i] = f32_to_bf16(src[(size_t)k * DD + n]);
}

// per-(dst, rel) in-degree counts
__global__ void count_edges(const int* __restrict__ ei,
                            const int* __restrict__ et,
                            float* __restrict__ cnt) {
    int i = blockIdx.x * blockDim.x + threadIdx.x;
    if (i >= BB * EE) return;
    int b = i / EE, e = i % EE;
    int dst = ei[(size_t)b * 2 * EE + EE + e];
    int r   = et[(size_t)b * EE + e];
    unsafeAtomicAdd(&cnt[((size_t)b * NN + dst) * RR + r], 1.0f);
}

// ------------------------------------------------------------------- GEMM
// Y[b,g] = xb[b] (N x D, bf16 row-major) @ wp[g] (D x D, packed as [out][k])
// g < RR  -> xw[b][g]            (raw f32)
// g == RR -> agg[b] = Y + bias   (root transform, initializes accumulator buf)
//
// 128 threads = 4 waves; block tile 128x128; wave tile 64x64 (16 acc tiles).
// B panel (128 rows x 32 K, 8 KB) is staged in LDS via the Tensor Data Mover
// with double buffering: prefetch of step i+1 overlaps the 16 WMMAs of step i.
__global__ __launch_bounds__(128)
void rgcn_gemm(const unsigned short* __restrict__ xb,
               const unsigned short* __restrict__ wp,
               float* __restrict__ xw,
               float* __restrict__ agg,
               const float* __restrict__ bias) {
    __shared__ alignas(16) unsigned short tileB[2][128 * 32];   // 2 x 8 KB

    const int lane    = threadIdx.x & 31;
    const int wave    = threadIdx.x >> 5;
    const int laneLow = lane & 15;
    const int laneHi  = lane >> 4;          // 0 or 1
    const int wm = wave >> 1, wn = wave & 1;

    const int b = blockIdx.z >> 2;
    const int g = blockIdx.z & 3;
    const int rbase  = blockIdx.x * 128 + wm * 64;   // M (node rows)
    const int cbase0 = blockIdx.y * 128;             // block N base (output cols)
    const int cwave  = wn * 64;                      // wave col offset in B tile

    const unsigned short* xb_b = xb + (size_t)b * NN * DD;
    const unsigned short* wp_g = wp + (size_t)g * DD * DD;

    // A fragments direct from global: lanes 0-15 row m K={0..7,16..23},
    // lanes 16-31 row m K={8..15,24..31} -> two b128 chunks 32 bytes apart.
    GQc pa[4];
#pragma unroll
    for (int mi = 0; mi < 4; ++mi)
        pa[mi] = (GQc)(const u32x4*)(xb_b + (size_t)(rbase + mi * 16 + laneLow) * DD + laneHi * 8);

    const unsigned ldsBase = (unsigned)(size_t)(LS16)&tileB[0][0];

    // Stage B tile for K-step `step` into buffer `buf`.
    auto stage = [&](int buf, int step) {
#if HAVE_TDM
        if (threadIdx.x < 32) {
            const unsigned ldsoff = ldsBase + (unsigned)buf * (128 * 32 * 2);
            const unsigned long long ga =
                (unsigned long long)(uintptr_t)(wp_g + (size_t)cbase0 * DD + (size_t)step * 32);
            // D# group 0: count=1 | lds_addr | global_addr(57b) | type=2
            u32x4 g0;
            g0[0] = 1u;
            g0[1] = ldsoff;
            g0[2] = (unsigned)ga;
            g0[3] = (unsigned)((ga >> 32) & 0x01FFFFFFu) | (2u << 30);
            // D# group 1: data_size=2B; tensor 768x768; tile 32x128; stride 768
            i32x8 g1;
            g1[0] = (int)(1u << 16);            // data_size = 1 (2 bytes)
            g1[1] = (int)(768u << 16);          // tensor_dim0 = 768 (bits 79:48 lo)
            g1[2] = (int)(768u << 16);          // tensor_dim1 = 768 (bits 111:80 lo)
            g1[3] = (int)(32u << 16);           // tile_dim0 = 32
            g1[4] = 128;                        // tile_dim1 = 128, tile_dim2 = 0
            g1[5] = 768;                        // tensor_dim0_stride = 768
            g1[6] = 0;                          // dim1_stride lo16 (unused, 2D tile)
            g1[7] = 9;                          // dim1_stride hi (768*768 >> 16)
            i32x4 z4 = 0;
#if __clang_major__ >= 23
            i32x8 z8 = 0;
            __builtin_amdgcn_tensor_load_to_lds(g0, g1, z4, z4, z8, 0);
#else
            __builtin_amdgcn_tensor_load_to_lds(g0, g1, z4, z4, 0);
#endif
        }
#else
        // Fallback: cooperative copy, one 64-byte row per thread.
        const int t = threadIdx.x;   // 0..127 rows
        GQc src = (GQc)(const u32x4*)(wp_g + (size_t)(cbase0 + t) * DD + (size_t)step * 32);
        LQp dst = (LQp)(u32x4*)&tileB[buf][t * 32];
        dst[0] = src[0]; dst[1] = src[1]; dst[2] = src[2]; dst[3] = src[3];
#endif
    };

    v8f acc[4][4] = {};

    stage(0, 0);

    for (int it = 0; it < NK; ++it) {
        const int cur = it & 1;
        if (it + 1 < NK) stage(cur ^ 1, it + 1);     // prefetch next B tile
#if HAVE_TDM
        if (threadIdx.x < 32) {
            if (it + 1 < NK) __builtin_amdgcn_s_wait_tensorcnt(1);  // cur done, next in flight
            else             __builtin_amdgcn_s_wait_tensorcnt(0);
        }
#endif
        __syncthreads();                             // publish buffer `cur`

        Frag fa[4], fb[4];
#pragma unroll
        for (int mi = 0; mi < 4; ++mi) {
            fa[mi].q[0] = pa[mi][0];
            fa[mi].q[1] = pa[mi][2];   // +32 bytes = +16 bf16 elements
            pa[mi] += 4;               // advance 32 K elements
        }
#pragma unroll
        for (int ni = 0; ni < 4; ++ni) {
            const int rowB = cwave + ni * 16 + laneLow;
            LQc lp = (LQc)(const u32x4*)&tileB[cur][rowB * 32 + laneHi * 8];
            fb[ni].q[0] = lp[0];
            fb[ni].q[1] = lp[2];
        }
#pragma unroll
        for (int mi = 0; mi < 4; ++mi)
#pragma unroll
            for (int ni = 0; ni < 4; ++ni)
                acc[mi][ni] = __builtin_amdgcn_wmma_f32_16x16x32_bf16(
                    false, fa[mi].v, false, fb[ni].v,
                    (short)0, acc[mi][ni], false, false);

        __syncthreads();                             // safe to overwrite `cur` next iter
    }

    // C/D layout: VGPR j -> row (j + 8*laneHi), col = laneLow
    if (g < RR) {
        GFp out = (GFp)(xw + ((size_t)b * RR + g) * NN * DD);
#pragma unroll
        for (int mi = 0; mi < 4; ++mi)
#pragma unroll
            for (int ni = 0; ni < 4; ++ni) {
                const int col = cbase0 + cwave + ni * 16 + laneLow;
#pragma unroll
                for (int j = 0; j < 8; ++j) {
                    const int row = rbase + mi * 16 + laneHi * 8 + j;
                    out[(size_t)row * DD + col] = acc[mi][ni][j];
                }
            }
    } else {
        GFp out = (GFp)(agg + (size_t)b * NN * DD);
        float bv[4];
#pragma unroll
        for (int ni = 0; ni < 4; ++ni) bv[ni] = bias[cbase0 + cwave + ni * 16 + laneLow];
#pragma unroll
        for (int mi = 0; mi < 4; ++mi)
#pragma unroll
            for (int ni = 0; ni < 4; ++ni) {
                const int col = cbase0 + cwave + ni * 16 + laneLow;
#pragma unroll
                for (int j = 0; j < 8; ++j) {
                    const int row = rbase + mi * 16 + laneHi * 8 + j;
                    out[(size_t)row * DD + col] = acc[mi][ni][j] + bv[ni];
                }
            }
    }
}

// ----------------------------------------------------------------- scatter
// one 192-thread block per edge; float4 gather + 4 f32 atomic adds per thread
__global__ __launch_bounds__(192)
void scatter_edges(const float* __restrict__ xw,
                   const int* __restrict__ ei,
                   const int* __restrict__ et,
                   const float* __restrict__ cnt,
                   float* __restrict__ agg) {
    const int idx = blockIdx.x;           // 0 .. B*E-1
    const int b = idx / EE, e = idx % EE;
    const int src = ei[(size_t)b * 2 * EE + e];
    const int dst = ei[(size_t)b * 2 * EE + EE + e];
    const int r   = et[(size_t)b * EE + e];
    const float c = cnt[((size_t)b * NN + dst) * RR + r];
    const float norm = 1.0f / fmaxf(c, 1.0f);

    GF4c msg = (GF4c)(const f32x4*)(xw + (((size_t)b * RR + r) * NN + src) * DD);
    float* outp = agg + ((size_t)b * NN + dst) * DD;

    const int t = threadIdx.x;            // DD/4 == 192
    f32x4 m = msg[t];
    unsafeAtomicAdd(&outp[t * 4 + 0], m.x * norm);
    unsafeAtomicAdd(&outp[t * 4 + 1], m.y * norm);
    unsafeAtomicAdd(&outp[t * 4 + 2], m.z * norm);
    unsafeAtomicAdd(&outp[t * 4 + 3], m.w * norm);
}

// ------------------------------------------------------------------ launch
extern "C" void kernel_launch(void* const* d_in, const int* in_sizes, int n_in,
                              void* d_out, int out_size, void* d_ws, size_t ws_size,
                              hipStream_t stream) {
    const float* x     = (const float*)d_in[0];
    const int*   ei    = (const int*)  d_in[1];
    const int*   et    = (const int*)  d_in[2];
    const float* W1    = (const float*)d_in[3];
    const float* root1 = (const float*)d_in[4];
    const float* b1    = (const float*)d_in[5];
    const float* W2    = (const float*)d_in[6];
    const float* root2 = (const float*)d_in[7];
    const float* b2    = (const float*)d_in[8];
    float* out = (float*)d_out;

    char* ws = (char*)d_ws;
    size_t off = 0;
    unsigned short* XB  = (unsigned short*)(ws + off); off += (size_t)BB * NN * DD * 2;  // 25.2 MB (reused as hb)
    unsigned short* WP1 = (unsigned short*)(ws + off); off += (size_t)NG * DD * DD * 2;  // 4.7 MB
    unsigned short* WP2 = (unsigned short*)(ws + off); off += (size_t)NG * DD * DD * 2;  // 4.7 MB
    float* XW  = (float*)(ws + off); off += (size_t)BB * RR * NN * DD * 4;               // 151 MB
    float* CNT = (float*)(ws + off); off += (size_t)BB * NN * RR * 4;                    // 0.2 MB
    float* H   = (float*)(ws + off); off += (size_t)BB * NN * DD * 4;                    // 50 MB

    const int n4 = BB * NN * DD / 4;
    const dim3 ggrid(NN / 128, DD / 128, BB * NG);   // 32 x 6 x 16

    // degree counts (edge structure is identical for both layers)
    zero_f32   <<<(BB * NN * RR + 255) / 256, 256, 0, stream>>>(CNT, BB * NN * RR);
    count_edges<<<(BB * EE + 255) / 256,      256, 0, stream>>>(ei, et, CNT);

    // weight packing (both layers)
    pack_weights<<<(NG * DD * DD + 255) / 256, 256, 0, stream>>>(W1, root1, WP1);
    pack_weights<<<(NG * DD * DD + 255) / 256, 256, 0, stream>>>(W2, root2, WP2);

    // ---- layer 1 ----
    convert_bf16<<<(n4 + 255) / 256, 256, 0, stream>>>(x, XB, n4, 0);
    rgcn_gemm   <<<ggrid, 128, 0, stream>>>(XB, WP1, XW, H, b1);      // xw + root-init of H
    scatter_edges<<<BB * EE, 192, 0, stream>>>(XW, ei, et, CNT, H);

    // ---- layer 2 (ReLU fused into conversion) ----
    convert_bf16<<<(n4 + 255) / 256, 256, 0, stream>>>(H, XB, n4, 1);
    rgcn_gemm   <<<ggrid, 128, 0, stream>>>(XB, WP2, XW, out, b2);    // xw + root-init of out
    scatter_edges<<<BB * EE, 192, 0, stream>>>(XW, ei, et, CNT, out);
}
